// SelfAttention_6270652252306
// MI455X (gfx1250) — matrix operand order repaired and checked by reading, not compiled, not run on hardware
//
#include <hip/hip_runtime.h>

// ---------------- CDNA5 wave32 WMMA types ----------------
typedef __attribute__((ext_vector_type(16))) __bf16 v16bf;
typedef __attribute__((ext_vector_type(8)))  float  v8f;
typedef __attribute__((ext_vector_type(4)))  float  v4f;
typedef __attribute__((ext_vector_type(4)))  int    v4i;
typedef __attribute__((ext_vector_type(4)))  __bf16 v4bf;
typedef __attribute__((ext_vector_type(2)))  __bf16 v2bf;

// ---------------- problem constants ----------------
constexpr int kB  = 8;     // batches
constexpr int kN  = 2048;  // sequence length
constexpr int kD  = 512;   // feature dim
constexpr int kBr = 64;    // query rows per workgroup (4 waves x 16)
constexpr int kBc = 32;    // keys per flash iteration (one WMMA K-step for PV)
constexpr int kWaves    = 4;
constexpr int kThreads  = 128;
constexpr int kKSteps   = kD / 32;   // 16 K-steps of the QK^T GEMM
constexpr int kDTiles   = kD / 16;   // 32 output column tiles
constexpr int kIters    = kN / kBc;  // 64 flash iterations

// LDS layout (bytes). kv row-major: [32][520] bf16 (pad 8 -> conflict-free
// B-frag reads). kvt transposed: [512][40] bf16 (stride 80B = 20 dw).
constexpr int kKVRStride = 520;
constexpr int kKVTStride = 40;
constexpr int kKVBytes   = 32 * kKVRStride * 2;   // 33280
constexpr int kKVTBytes  = 512 * kKVTStride * 2;  // 40960
constexpr int kOffKV   = 0;                        // two buffers back to back
constexpr int kOffKVT  = 2 * kKVBytes;             // two buffers back to back
constexpr int kOffPBuf = 2 * kKVBytes + 2 * kKVTBytes;           // 148480
constexpr int kPBufBytesPerWave = 16 * kKVTStride * 2;            // 1280
constexpr int kSmemBytes = kOffPBuf + kWaves * kPBufBytesPerWave; // 153600

// ---------------- helpers ----------------
__device__ inline v8f vzero8() {
  v8f z;
#pragma unroll
  for (int i = 0; i < 8; ++i) z[i] = 0.0f;
  return z;
}

// 16 contiguous bf16 from LDS (two ds_load_b128)
__device__ inline v16bf lds_ld16(const __bf16* p) {
  union { v4i i[2]; v16bf b; } u;
  u.i[0] = *(const v4i*)(p);
  u.i[1] = *(const v4i*)(p + 8);
  return u.b;
}
// two separated 8-element (16B) chunks -> one A-fragment register group
__device__ inline v16bf lds_ld2x8(const __bf16* p0, const __bf16* p1) {
  union { v4i i[2]; v16bf b; } u;
  u.i[0] = *(const v4i*)(p0);
  u.i[1] = *(const v4i*)(p1);
  return u.b;
}

__device__ inline void prefetch_tile(const float* __restrict__ xb, int jtile, int tid) {
  const char* base = (const char*)(xb + (size_t)jtile * kBc * kD);
#pragma unroll
  for (int i = 0; i < 4; ++i)
    __builtin_prefetch(base + (size_t)(tid * 4 + i) * 128, 0, 1);  // global_prefetch_b8
}

// Load one 32x512 fp32 tile of x, convert to bf16, store row-major + transposed.
// Each thread owns a PAIR of key rows (2*kp, 2*kp+1) over a 64-wide d chunk so
// the transposed copy can be written as packed bf16x2 ds_store_b32 (key pairs
// share a dword in kvt), instead of scalar b16 stores.
__device__ inline void fill_tile(const float* __restrict__ xb, int jtile,
                                 __bf16* kvr, __bf16* kvt, int tid) {
  const int kp    = tid & 15;   // key pair index
  const int chunk = tid >> 4;   // 0..7 -> d range [chunk*64, chunk*64+64)
  const int r0 = 2 * kp;
  const int r1 = 2 * kp + 1;
  const int dbase = chunk * 64;
  const float* s0p = xb + (size_t)(jtile * kBc + r0) * kD + dbase;
  const float* s1p = xb + (size_t)(jtile * kBc + r1) * kD + dbase;
#pragma unroll
  for (int i = 0; i < 16; ++i) {
    v4f f0 = *(const v4f*)(s0p + i * 4);
    v4f f1 = *(const v4f*)(s1p + i * 4);
    const int d = dbase + i * 4;
    v4bf p0, p1;
#pragma unroll
    for (int q = 0; q < 4; ++q) { p0[q] = (__bf16)f0[q]; p1[q] = (__bf16)f1[q]; }
    *(v4bf*)(kvr + r0 * kKVRStride + d) = p0;   // ds_store_b64
    *(v4bf*)(kvr + r1 * kKVRStride + d) = p1;   // ds_store_b64
#pragma unroll
    for (int q = 0; q < 4; ++q) {               // transposed, pair-packed b32
      v2bf pr; pr[0] = p0[q]; pr[1] = p1[q];
      *(v2bf*)(kvt + (d + q) * kKVTStride + r0) = pr;
    }
  }
}

// ---------------- fused flash self-attention ----------------
__global__ __launch_bounds__(kThreads, 1)
void attn_flash_bf16(const float* __restrict__ x, float* __restrict__ out) {
  extern __shared__ char smem[];
  const int tid  = threadIdx.x;
  const int wave = tid >> 5;
  const int lane = tid & 31;
  const int hf   = lane >> 4;   // half-wave index (0/1)
  const int l16  = lane & 15;

  const int blk  = blockIdx.x;          // 256 blocks
  const int b    = blk >> 5;            // batch
  const int qblk = (blk & 31) * kBr;    // query row base of this workgroup
  const float* __restrict__ xb = x   + (size_t)b * kN * kD;
  float* __restrict__       ob = out + (size_t)b * kN * kD;

  __bf16* pb = (__bf16*)(smem + kOffPBuf + wave * kPBufBytesPerWave);

  // ---- preload Q tile as 16 bf16 A-fragments (lane: row = l16; element e ->
  // k_local = e + 16*hf, contiguous order matching the B-fragments) ----
  const int qrowA = qblk + wave * 16 + l16;
  v16bf qa[kKSteps];
#pragma unroll
  for (int ks = 0; ks < kKSteps; ++ks) {
    const float* src = xb + (size_t)qrowA * kD + ks * 32 + hf * 16;
    v4f f0 = *(const v4f*)(src);
    v4f f1 = *(const v4f*)(src + 4);
    v4f f2 = *(const v4f*)(src + 8);
    v4f f3 = *(const v4f*)(src + 12);
    v16bf a;
#pragma unroll
    for (int i = 0; i < 4; ++i) {
      a[i]      = (__bf16)f0[i];
      a[4 + i]  = (__bf16)f1[i];
      a[8 + i]  = (__bf16)f2[i];
      a[12 + i] = (__bf16)f3[i];
    }
    qa[ks] = a;
  }

  // ---- flash state ----
  float mrow[8], lsum[8];
#pragma unroll
  for (int v = 0; v < 8; ++v) { mrow[v] = -__builtin_inff(); lsum[v] = 0.0f; }
  v8f o[kDTiles];
#pragma unroll
  for (int t = 0; t < kDTiles; ++t) o[t] = vzero8();

  prefetch_tile(xb, 1, tid);
  fill_tile(xb, 0,
            (__bf16*)(smem + kOffKV),
            (__bf16*)(smem + kOffKVT), tid);
  __syncthreads();

  for (int j = 0; j < kIters; ++j) {
    const int cur = j & 1;
    const int nxt = cur ^ 1;
    if (j + 2 < kIters) prefetch_tile(xb, j + 2, tid);
    if (j + 1 < kIters)
      fill_tile(xb, j + 1,
                (__bf16*)(smem + kOffKV  + nxt * kKVBytes),
                (__bf16*)(smem + kOffKVT + nxt * kKVTBytes), tid);

    // ---- S = Q_tile (16x512) * K_blk^T -> two 16x16 f32 tiles ----
    // Software-pipelined: B-fragments for step ks+1 are in flight while the
    // WMMAs for step ks execute (4 ds_load_b128 outstanding).
    const __bf16* kvb  = (const __bf16*)(smem + kOffKV + cur * kKVBytes);
    const __bf16* kvb0 = kvb + (l16)      * kKVRStride + hf * 16;
    const __bf16* kvb1 = kvb + (16 + l16) * kKVRStride + hf * 16;
    v8f s0 = vzero8(), s1 = vzero8();
    v16bf b0 = lds_ld16(kvb0);
    v16bf b1 = lds_ld16(kvb1);
#pragma unroll
    for (int ks = 0; ks < kKSteps; ++ks) {
      v16bf nb0 = b0, nb1 = b1;
      if (ks + 1 < kKSteps) {
        nb0 = lds_ld16(kvb0 + (ks + 1) * 32);
        nb1 = lds_ld16(kvb1 + (ks + 1) * 32);
      }
      s0 = __builtin_amdgcn_wmma_f32_16x16x32_bf16(false, qa[ks], false, b0,
                                                   (short)0, s0, false, false);
      s1 = __builtin_amdgcn_wmma_f32_16x16x32_bf16(false, qa[ks], false, b1,
                                                   (short)0, s1, false, false);
      b0 = nb0; b1 = nb1;
    }

    // ---- streaming softmax over the 32 new keys ----
    float scale[8];
#pragma unroll
    for (int v = 0; v < 8; ++v) {
      float m = fmaxf(s0[v], s1[v]);
      m = fmaxf(m, __shfl_xor(m, 1, 32));
      m = fmaxf(m, __shfl_xor(m, 2, 32));
      m = fmaxf(m, __shfl_xor(m, 4, 32));
      m = fmaxf(m, __shfl_xor(m, 8, 32));
      const float mn = fmaxf(mrow[v], m);
      scale[v] = __expf(mrow[v] - mn);
      mrow[v]  = mn;
      const float p0 = __expf(s0[v] - mn);
      const float p1 = __expf(s1[v] - mn);
      s0[v] = p0; s1[v] = p1;
      float rs = p0 + p1;
      rs += __shfl_xor(rs, 1, 32);
      rs += __shfl_xor(rs, 2, 32);
      rs += __shfl_xor(rs, 4, 32);
      rs += __shfl_xor(rs, 8, 32);
      lsum[v] = lsum[v] * scale[v] + rs;
    }
#pragma unroll
    for (int t = 0; t < kDTiles; ++t)
#pragma unroll
      for (int v = 0; v < 8; ++v) o[t][v] *= scale[v];

    // ---- re-layout P: C-layout -> A-fragment via per-wave LDS scratch ----
#pragma unroll
    for (int v = 0; v < 8; ++v) {
      const int r = v + 8 * hf;                 // C-layout row
      pb[r * kKVTStride + l16]      = (__bf16)s0[v];
      pb[r * kKVTStride + 16 + l16] = (__bf16)s1[v];
    }
    const __bf16* prow = pb + l16 * kKVTStride;
    v16bf pa = lds_ld2x8(prow + hf * 16, prow + hf * 16 + 8);

    // ---- O += P (16x32) * V_blk (32x512), depth-2 pipelined B-fragments ----
    const __bf16* vtb = (const __bf16*)(smem + kOffKVT + cur * kKVTBytes);
    const __bf16* vt0 = vtb + l16 * kKVTStride + hf * 16;
    v16bf bv0 = lds_ld16(vt0);
    v16bf bv1 = lds_ld16(vt0 + 16 * kKVTStride);
#pragma unroll
    for (int t = 0; t < kDTiles; ++t) {
      v16bf nb = bv1;
      if (t + 2 < kDTiles) nb = lds_ld16(vt0 + (t + 2) * 16 * kKVTStride);
      o[t] = __builtin_amdgcn_wmma_f32_16x16x32_bf16(false, pa, false, bv0,
                                                     (short)0, o[t], false, false);
      bv0 = bv1; bv1 = nb;
    }

    __syncthreads();
  }

  // ---- epilogue: out = (O / lsum) .* x ----
  float inv[8];
#pragma unroll
  for (int v = 0; v < 8; ++v) inv[v] = 1.0f / lsum[v];
#pragma unroll
  for (int t = 0; t < kDTiles; ++t) {
#pragma unroll
    for (int v = 0; v < 8; ++v) {
      const int r = qblk + wave * 16 + v + 8 * hf;  // C-layout row mapping
      const int c = t * 16 + l16;
      const size_t idx = (size_t)r * kD + c;
      ob[idx] = o[t][v] * inv[v] * xb[idx];
    }
  }
}

// ---------------- host entry ----------------
extern "C" void kernel_launch(void* const* d_in, const int* in_sizes, int n_in,
                              void* d_out, int out_size, void* d_ws, size_t ws_size,
                              hipStream_t stream) {
  (void)in_sizes; (void)n_in; (void)d_ws; (void)ws_size; (void)out_size;
  const float* x = (const float*)d_in[0];
  float* out = (float*)d_out;
  dim3 grid(kB * (kN / kBr));   // 256 workgroups
  dim3 block(kThreads);         // 128 threads = 4 waves (wave32)
  attn_flash_bf16<<<grid, block, kSmemBytes, stream>>>(x, out);
}
